// Router_42082089566761
// MI455X (gfx1250) — matrix operand order repaired and checked by reading, not compiled, not run on hardware
//
#include <hip/hip_runtime.h>
#include <hip/hip_bf16.h>
#include <math.h>

typedef __attribute__((ext_vector_type(16))) __bf16 v16bf;
typedef __attribute__((ext_vector_type(8)))  __bf16 v8bf;
typedef __attribute__((ext_vector_type(4)))  __bf16 v4bf;
typedef __attribute__((ext_vector_type(8)))  float  v8f;

#define D_MODEL       2048
#define NUM_EXPERTS   64
#define WROW_BF       2056                 // padded W row in bf16 elems (2048 + 8)
#define WROW_BYTES    (WROW_BF * 2)        // 4112 B; 4-bank skew per expert row
#define LOGITS_STRIDE 68                   // floats per token row (16B aligned, bank-skewed)
#define TOK_PER_WG    128
#define THREADS       256
#define WBF_BYTES     ((size_t)NUM_EXPERTS * D_MODEL * 2)   // 256 KB

// ---------- One-shot W converter: fp32 -> bf16 into workspace ----------
__global__ __launch_bounds__(THREADS)
void wcvt_kernel(const float* __restrict__ W, __bf16* __restrict__ wbf) {
    const int i = (blockIdx.x * THREADS + threadIdx.x) * 4;
    const float4 w4 = *(const float4*)(W + i);
    v4bf h;
    h[0] = (__bf16)w4.x; h[1] = (__bf16)w4.y;
    h[2] = (__bf16)w4.z; h[3] = (__bf16)w4.w;
    *(v4bf*)(wbf + i) = h;
}

__global__ __launch_bounds__(THREADS)
void router_wmma_kernel(const float* __restrict__ x,
                        const float* __restrict__ W,
                        const __bf16* __restrict__ Wbf,   // may be null
                        float* __restrict__ gates_out,
                        float* __restrict__ idx_out)
{
    extern __shared__ char smem[];
    float* Llds = (float*)(smem + (size_t)NUM_EXPERTS * WROW_BYTES);

    const int tid = threadIdx.x;

    // ---------- Stage W into LDS (padded rows) ----------
    if (Wbf != nullptr) {
        // bf16 source in workspace: pure 16B copies (L2-resident).
        #pragma unroll 4
        for (int it = 0; it < (NUM_EXPERTS * D_MODEL) / (THREADS * 8); ++it) {
            const int i = (it * THREADS + tid) * 8;       // bf16 elem index
            const int e = i >> 11;
            const int d = i & (D_MODEL - 1);
            *(v8bf*)(smem + (size_t)e * WROW_BYTES + (size_t)d * 2) =
                *(const v8bf*)(Wbf + i);
        }
    } else {
        // Fallback: convert fp32 W in-kernel.
        #pragma unroll 4
        for (int it = 0; it < (NUM_EXPERTS * D_MODEL) / (THREADS * 4); ++it) {
            const int i = (it * THREADS + tid) * 4;
            const float4 w4 = *(const float4*)(W + i);
            const int e = i >> 11;
            const int d = i & (D_MODEL - 1);
            v4bf h;
            h[0] = (__bf16)w4.x; h[1] = (__bf16)w4.y;
            h[2] = (__bf16)w4.z; h[3] = (__bf16)w4.w;
            *(v4bf*)(smem + (size_t)e * WROW_BYTES + (size_t)d * 2) = h;
        }
    }
    __syncthreads();

    // ---------- WMMA main loop: 16 tokens/wave, 64 experts in 4 N-tiles ----------
    const int wave = tid >> 5;
    const int lane = tid & 31;
    const int half = lane >> 4;   // 0: lanes 0-15, 1: lanes 16-31
    const int r    = lane & 15;

    const int tokBase = blockIdx.x * TOK_PER_WG + wave * 16;
    // A fragment (16x32 bf16), ISA 7.12.2: lanes 0-15 hold K {0..7,16..23} of
    // row r; lanes 16-31 hold K {8..15,24..31}.
    const float* aPtr = x + (size_t)(tokBase + r) * D_MODEL + half * 8;

    v8f acc[4] = {};   // f32 accumulators: 4 tiles of 16 experts

    // Per-tile LDS BYTE OFFSETS as induction variables (+64 B per K sub-step).
    // The pointer is re-derived from `smem` at every access so the compiler
    // keeps the LDS address space (ds_load_b128, NOT flat_load).
    unsigned bOff[4];
    #pragma unroll
    for (int t = 0; t < 4; ++t)
        bOff[t] = (unsigned)((t * 16 + r) * WROW_BYTES + half * 16);

    // Double-buffered A tiles in FIXED registers (no loop-carried copies).
    float4 A0[4], A1[4];

    #define LOAD_A(BUF, OFF) do {                                   \
        const float* _p = aPtr + (OFF);                             \
        (BUF)[0] = *(const float4*)(_p + 0);                        \
        (BUF)[1] = *(const float4*)(_p + 4);                        \
        (BUF)[2] = *(const float4*)(_p + 16);                       \
        (BUF)[3] = *(const float4*)(_p + 20);                       \
    } while (0)

    #define CVT_A(DST, BUF) do {                                    \
        const float _fv[16] = {                                     \
            (BUF)[0].x, (BUF)[0].y, (BUF)[0].z, (BUF)[0].w,         \
            (BUF)[1].x, (BUF)[1].y, (BUF)[1].z, (BUF)[1].w,         \
            (BUF)[2].x, (BUF)[2].y, (BUF)[2].z, (BUF)[2].w,         \
            (BUF)[3].x, (BUF)[3].y, (BUF)[3].z, (BUF)[3].w };       \
        _Pragma("unroll")                                           \
        for (int _i = 0; _i < 16; ++_i) (DST)[_i] = (__bf16)_fv[_i];\
    } while (0)

    #define MMA_STEP(AV) do {                                       \
        v16bf _bf[4];                                               \
        _Pragma("unroll")                                           \
        for (int _t = 0; _t < 4; ++_t) {                            \
            const v8bf _b0 = *(const v8bf*)(smem + bOff[_t]);       \
            const v8bf _b1 = *(const v8bf*)(smem + bOff[_t] + 32);  \
            _Pragma("unroll")                                       \
            for (int _i = 0; _i < 8; ++_i) {                        \
                _bf[_t][_i] = _b0[_i]; _bf[_t][8 + _i] = _b1[_i];   \
            }                                                       \
            bOff[_t] += 64;                                         \
        }                                                           \
        _Pragma("unroll")                                           \
        for (int _t = 0; _t < 4; ++_t) {                            \
            acc[_t] = __builtin_amdgcn_wmma_f32_16x16x32_bf16(      \
                false, (AV), false, _bf[_t], (short)0, acc[_t], false, false); \
        }                                                           \
    } while (0)

    LOAD_A(A0, 0);
    for (int k = 0; k < D_MODEL; k += 64) {
        // ---- sub-step 0: consume A0, refill A1 for k+32 ----
        v16bf a0;
        CVT_A(a0, A0);
        LOAD_A(A1, k + 32);                       // always in-bounds
        // GL2 prefetch 3 steps ahead; offset wrapped in-row (always valid).
        __builtin_prefetch(aPtr + ((k + 96) & (D_MODEL - 1)), 0, 3);
        MMA_STEP(a0);

        // ---- sub-step 1: consume A1, refill A0 for k+64 (wraps on last) ----
        v16bf a1;
        CVT_A(a1, A1);
        LOAD_A(A0, (k + 64) & (D_MODEL - 1));     // last refill wraps to 0 (unused)
        __builtin_prefetch(aPtr + ((k + 128) & (D_MODEL - 1)), 0, 3);
        MMA_STEP(a1);
    }

    #undef LOAD_A
    #undef CVT_A
    #undef MMA_STEP

    // ---------- Scatter logits to LDS (C layout: lane=col, vgpr=row) ----------
    {
        const int m0 = half * 8;
        #pragma unroll
        for (int t = 0; t < 4; ++t) {
            const int e = t * 16 + r;
            #pragma unroll
            for (int m = 0; m < 8; ++m) {
                Llds[(wave * 16 + m0 + m) * LOGITS_STRIDE + e] = acc[t][m];
            }
        }
    }
    __syncthreads();

    // ---------- Epilogue: branchless top-2 of 64 logits per token ----------
    // softmax + top-2 + renorm collapses to g0 = 1/(1+exp(l1-l0)).
    if (tid < TOK_PER_WG) {
        const float* lrow = Llds + tid * LOGITS_STRIDE;
        float l0 = -INFINITY, l1 = -INFINITY;
        int   i0 = 0, i1 = 0;
        #pragma unroll
        for (int e = 0; e < NUM_EXPERTS; e += 4) {
            const float4 v = *(const float4*)(lrow + e);
            const float vv[4] = { v.x, v.y, v.z, v.w };
            #pragma unroll
            for (int j = 0; j < 4; ++j) {
                const float val = vv[j];
                const int   idx = e + j;
                const bool gt0 = val > l0;
                const bool gt1 = val > l1;
                const float nl1 = gt0 ? l0 : (gt1 ? val : l1);
                const int   ni1 = gt0 ? i0 : (gt1 ? idx : i1);
                l0 = gt0 ? val : l0;
                i0 = gt0 ? idx : i0;
                l1 = nl1;
                i1 = ni1;
            }
        }
        const float g0 = 1.0f / (1.0f + __expf(l1 - l0));
        const size_t tok = (size_t)blockIdx.x * TOK_PER_WG + tid;
        gates_out[tok * 2]     = g0;
        gates_out[tok * 2 + 1] = 1.0f - g0;
        idx_out[tok * 2]       = (float)i0;
        idx_out[tok * 2 + 1]   = (float)i1;
    }
}

extern "C" void kernel_launch(void* const* d_in, const int* in_sizes, int n_in,
                              void* d_out, int out_size, void* d_ws, size_t ws_size,
                              hipStream_t stream) {
    const float* x = (const float*)d_in[0];
    const float* W = (const float*)d_in[1];
    const int T = in_sizes[0] / D_MODEL;          // total tokens (32768)

    float* gates = (float*)d_out;                 // [T, 2] gates
    float* idxf  = (float*)d_out + (size_t)T * 2; // [T, 2] indices (as float)

    // Pre-convert W to bf16 in workspace if it fits (host-constant decision).
    __bf16* wbf = nullptr;
    if (ws_size >= WBF_BYTES) {
        wbf = (__bf16*)d_ws;
        const int n = NUM_EXPERTS * D_MODEL;                  // 131072
        wcvt_kernel<<<n / (THREADS * 4), THREADS, 0, stream>>>(W, wbf);
    }

    const int blocks = (T + TOK_PER_WG - 1) / TOK_PER_WG;
    const size_t shmem = (size_t)NUM_EXPERTS * WROW_BYTES
                       + (size_t)TOK_PER_WG * LOGITS_STRIDE * sizeof(float);

    router_wmma_kernel<<<blocks, THREADS, shmem, stream>>>(x, W, wbf, gates, idxf);
}
